// StructureAdapter_3238405341430
// MI455X (gfx1250) — matrix operand order repaired and checked by reading, not compiled, not run on hardware
//
#include <hip/hip_runtime.h>
#include <hip/hip_bf16.h>

// ---------------------------------------------------------------------------
// MI455X (gfx1250) implementation.
//  - GEMMs: 64x64 workgroup tile, double-buffered LDS staging via
//    global_load_async_to_lds_b128 (+ s_wait_asynccnt), WMMA f16->f32.
//  - Attention fused per 16-row strip: WMMA logits -> 128KB fp32 LDS strip
//    (CDNA5: 320KB/WGP) -> softmax -> fp32 attention out -> WMMA attn@V.
//  - vT stored [B,H,D,S] so attn@V B-fragments are contiguous b128 loads.
// ---------------------------------------------------------------------------

typedef __attribute__((ext_vector_type(16))) _Float16 v16h;
typedef __attribute__((ext_vector_type(8)))  _Float16 v8h;
typedef __attribute__((ext_vector_type(8)))  float    v8f;

#define NHEADS 12
#define HDIM   64
#define BATCH  2
#define SLEN   2048
#define ROWS   (BATCH * SLEN)   // 4096
#define EMB    768
#define PLMD   1280
#define FFD    640

// ---- fragment loaders (ISA 7.12.2 layouts) --------------------------------
// A 16x32 f16: lane m = lane&15, g = lane>>4.
//   halves 0..7  -> K = k0 + 8*g + i ; halves 8..15 -> K = k0 + 16 + 8*g + i
static __device__ __forceinline__ v16h load_afrag(const _Float16* row, int k0, int g) {
    v8h lo = *(const v8h*)(row + k0 + 8 * g);
    v8h hi = *(const v8h*)(row + k0 + 16 + 8 * g);
    v16h a;
#pragma unroll
    for (int i = 0; i < 8; ++i) { a[i] = lo[i]; a[8 + i] = hi[i]; }
    return a;
}
// B 32x16 f16: lane n = lane&15, g = lane>>4.
//   halves 0..7  -> K = k0 + 8*g + i ; halves 8..15 -> K = k0 + 16 + 8*g + i
static __device__ __forceinline__ v16h load_bfrag(const _Float16* row, int k0, int g) {
    v8h lo = *(const v8h*)(row + k0 + 8 * g);
    v8h hi = *(const v8h*)(row + k0 + 16 + 8 * g);
    v16h b;
#pragma unroll
    for (int i = 0; i < 8; ++i) { b[i] = lo[i]; b[8 + i] = hi[i]; }
    return b;
}

// ---- CDNA5 async global->LDS copy (one b128 per lane, ASYNCcnt-tracked) ---
static __device__ __forceinline__ void async_copy_b128(const void* gsrc, void* lds_dst) {
    unsigned ldsa = (unsigned)(size_t)lds_dst;   // low 32 bits = LDS offset
    asm volatile("global_load_async_to_lds_b128 %0, %1, off"
                 :: "v"(ldsa), "v"(gsrc) : "memory");
}
static __device__ __forceinline__ void wait_async0() {
    asm volatile("s_wait_asynccnt 0x0" ::: "memory");
}

// ---- elementwise f32 -> f16 ----------------------------------------------
__global__ void k_cvt(const float* __restrict__ s, _Float16* __restrict__ d, int n) {
    int i = blockIdx.x * 256 + threadIdx.x;
    if (i < n) d[i] = (_Float16)s[i];
}

// ---- WMMA GEMM: C[M,N] = A[M,K] @ B[N,K]^T + bias -------------------------
// 256 threads, 64x64 output tile, LDS double-buffered K-slabs of 32.
// epi: 0 = bias only; 1 = bias + resid[M,N]; 2 = bias + exact erf-GELU
__global__ __launch_bounds__(256) void k_gemm(
    const _Float16* __restrict__ A, const _Float16* __restrict__ Bm,
    const float* __restrict__ bias, const float* __restrict__ resid,
    float* __restrict__ outF, _Float16* __restrict__ outH,
    int M, int N, int K, int epi)
{
    __shared__ __align__(16) _Float16 As[2][64 * 32];   // 4 KB per buffer
    __shared__ __align__(16) _Float16 Bs[2][64 * 32];

    const int t    = threadIdx.x;
    const int lane = t & 31;
    const int wave = t >> 5;          // 8 waves
    const int ln   = lane & 15;
    const int g    = lane >> 4;
    const int wr   = wave >> 2;       // 0..1 : 32-row band
    const int wc   = wave & 3;        // 0..3 : 16-col slice
    const int tm   = blockIdx.y;      // 64-row tile
    const int tn   = blockIdx.x;      // 64-col tile

    // per-thread staging slot: one b128 per lane fills a 64x32 f16 tile
    const int lrow = t >> 2;          // 0..63
    const int lcg  = (t & 3) * 8;     // 0,8,16,24 halves
    const _Float16* aG = A  + (size_t)(tm * 64 + lrow) * K + lcg;
    const _Float16* bG = Bm + (size_t)(tn * 64 + lrow) * K + lcg;
    const int lslot = lrow * 32 + lcg;

    // prologue: stage slab 0
    async_copy_b128(aG, &As[0][lslot]);
    async_copy_b128(bG, &Bs[0][lslot]);

    const int nk = K / 32;
    v8f acc0 = {}, acc1 = {};
    for (int ks = 0; ks < nk; ++ks) {
        wait_async0();
        __syncthreads();              // slab ks ready everywhere; prev compute done
        const int cur = ks & 1;
        if (ks + 1 < nk) {            // overlap: stage next slab into other buffer
            const int nb = cur ^ 1;
            async_copy_b128(aG + (ks + 1) * 32, &As[nb][lslot]);
            async_copy_b128(bG + (ks + 1) * 32, &Bs[nb][lslot]);
        }
        const _Float16* arow0 = &As[cur][(wr * 32 + ln) * 32];
        const _Float16* arow1 = &As[cur][(wr * 32 + 16 + ln) * 32];
        const _Float16* brow  = &Bs[cur][(wc * 16 + ln) * 32];
        v16h b  = load_bfrag(brow, 0, g);
        v16h a0 = load_afrag(arow0, 0, g);
        v16h a1 = load_afrag(arow1, 0, g);
        acc0 = __builtin_amdgcn_wmma_f32_16x16x32_f16(false, a0, false, b,
                                                      (short)0, acc0, false, false);
        acc1 = __builtin_amdgcn_wmma_f32_16x16x32_f16(false, a1, false, b,
                                                      (short)0, acc1, false, false);
    }

    const int   col = tn * 64 + wc * 16 + ln;
    const float bv  = bias[col];
#pragma unroll
    for (int r = 0; r < 8; ++r) {
        const int row0 = tm * 64 + wr * 32 + 8 * g + r;   // D: vgpr r -> M = r + 8g
#pragma unroll
        for (int h = 0; h < 2; ++h) {
            const int row = row0 + 16 * h;
            const size_t o = (size_t)row * N + col;
            float v = (h ? acc1[r] : acc0[r]) + bv;
            if (epi == 1) v += resid[o];
            if (epi == 2) v = 0.5f * v * (1.0f + erff(v * 0.70710678118654752f));
            if (outF) outF[o] = v;
            if (outH) outH[o] = (_Float16)v;
        }
    }
}

// ---- RoPE + relayout: qf/kf/vf [B,S,768] f32 -> qh/kh [B,H,S,D] f16, vT [B,H,D,S]
__global__ void k_rope(const float* __restrict__ qf, const float* __restrict__ kf,
                       const float* __restrict__ vf, _Float16* __restrict__ qh,
                       _Float16* __restrict__ kh, _Float16* __restrict__ vT)
{
    int idx = blockIdx.x * 256 + threadIdx.x;       // over ROWS*768
    if (idx >= ROWS * EMB) return;
    int d = idx & 63;
    int h = (idx >> 6) % NHEADS;
    int s = (idx / EMB) & (SLEN - 1);
    int b = idx / (EMB * SLEN);

    int   i    = d & 31;
    float freq = __powf(10000.0f, -(float)i * (1.0f / 32.0f));
    float ang  = (float)s * freq;
    float c    = cosf(ang);
    float sn   = sinf(ang);

    float q0 = qf[idx], kk0 = kf[idx];
    float qr = (d < 32) ? -qf[idx + 32] : qf[idx - 32];
    float kr = (d < 32) ? -kf[idx + 32] : kf[idx - 32];

    size_t o = ((size_t)(b * NHEADS + h) * SLEN + s) * HDIM + d;
    qh[o] = (_Float16)((q0 * c + qr * sn) * 0.125f);   // fold 1/sqrt(64) into q
    kh[o] = (_Float16)(kk0 * c + kr * sn);
    vT[((size_t)(b * NHEADS + h) * HDIM + d) * SLEN + s] = (_Float16)vf[idx];
}

// ---- fused attention: one block = 16 query rows of one (b,h) --------------
__global__ __launch_bounds__(256) void k_attn(
    const _Float16* __restrict__ qh, const _Float16* __restrict__ kh,
    const _Float16* __restrict__ vT, float* __restrict__ attn,
    _Float16* __restrict__ valsh)
{
    __shared__ __align__(16) float probs[16 * SLEN];     // 128 KB fp32 strip
    __shared__ __align__(16) float red[2 * 16 * HDIM];   // split-K reduce (8 KB)
    __shared__ float rowsum[16];

    const int rt   = blockIdx.x;                // 16-row tile of queries
    const int hidx = blockIdx.y;
    const int bb   = blockIdx.z;
    const int bh   = bb * NHEADS + hidx;
    const int lane = threadIdx.x & 31;
    const int wave = threadIdx.x >> 5;          // 8 waves
    const int ln   = lane & 15;
    const int g    = lane >> 4;

    const _Float16* qbase = qh + ((size_t)bh * SLEN + rt * 16) * HDIM;
    const _Float16* kbase = kh + (size_t)bh * SLEN * HDIM;
    const _Float16* vbase = vT + (size_t)bh * HDIM * SLEN;

    // Q A-fragments (K = 64 -> 2 fragments), shared by all key tiles
    const _Float16* arow = qbase + ln * HDIM;
    v16h aq0 = load_afrag(arow, 0, g);
    v16h aq1 = load_afrag(arow, 32, g);

    // Phase 1: logits (already scaled via q) into LDS
    for (int ct = wave; ct < SLEN / 16; ct += 8) {
        const _Float16* brow = kbase + (size_t)(ct * 16 + ln) * HDIM;
        v16h b0 = load_bfrag(brow, 0, g);
        v16h b1 = load_bfrag(brow, 32, g);
        v8f acc = {};
        acc = __builtin_amdgcn_wmma_f32_16x16x32_f16(false, aq0, false, b0,
                                                     (short)0, acc, false, false);
        acc = __builtin_amdgcn_wmma_f32_16x16x32_f16(false, aq1, false, b1,
                                                     (short)0, acc, false, false);
#pragma unroll
        for (int r = 0; r < 8; ++r)
            probs[(8 * g + r) * SLEN + ct * 16 + ln] = acc[r];
    }
    __syncthreads();

    // Phase 2: softmax stats; normalized fp32 attention to d_out; keep
    // UNnormalized exp() in LDS (1/sum folded into phase-3 epilogue).
    for (int r = wave; r < 16; r += 8) {
        float* row = probs + r * SLEN;
        float mx = -3.4e38f;
        for (int j = lane; j < SLEN; j += 32) mx = fmaxf(mx, row[j]);
#pragma unroll
        for (int off = 16; off; off >>= 1) mx = fmaxf(mx, __shfl_xor(mx, off, 32));
        float sm = 0.f;
        for (int j = lane; j < SLEN; j += 32) {
            float e = expf(row[j] - mx);
            row[j] = e;
            sm += e;
        }
#pragma unroll
        for (int off = 16; off; off >>= 1) sm += __shfl_xor(sm, off, 32);
        if (lane == 0) rowsum[r] = sm;
        float inv = 1.0f / sm;
        float* gout = attn + ((size_t)bh * SLEN + rt * 16 + r) * SLEN;
        for (int j = lane; j < SLEN; j += 32) gout[j] = row[j] * inv;
    }
    __syncthreads();

    // Phase 3: vals = exp_probs @ V, split-K across wave pairs
    const int nt  = wave & 3;     // 16-wide slice of head dim
    const int ksp = wave >> 2;    // K half: [0,1024) or [1024,2048)
    const float*    prow = probs + ln * SLEN;                      // A row m = ln
    const _Float16* vrow = vbase + (size_t)(nt * 16 + ln) * SLEN;  // vT row d
    v8f acc = {};
    for (int k0 = ksp * (SLEN / 2); k0 < (ksp + 1) * (SLEN / 2); k0 += 32) {
        const float* pl = prow + k0 + 8 * g;
        const float* ph = prow + k0 + 16 + 8 * g;
        v16h a;
#pragma unroll
        for (int i = 0; i < 8; ++i) { a[i] = (_Float16)pl[i]; a[8 + i] = (_Float16)ph[i]; }
        v16h b = load_bfrag(vrow, k0, g);
        acc = __builtin_amdgcn_wmma_f32_16x16x32_f16(false, a, false, b,
                                                     (short)0, acc, false, false);
    }
#pragma unroll
    for (int r = 0; r < 8; ++r)
        red[(ksp * 16 + 8 * g + r) * HDIM + nt * 16 + ln] = acc[r];
    __syncthreads();

    // combine split-K halves, apply 1/rowsum, write vals f16 as [B,S,H*D]
    for (int e = threadIdx.x; e < 16 * HDIM; e += 256) {
        int r = e >> 6, c = e & 63;
        float v = (red[r * HDIM + c] + red[(16 + r) * HDIM + c]) / rowsum[r];
        valsh[((size_t)(bb * SLEN + rt * 16 + r)) * EMB + hidx * HDIM + c] = (_Float16)v;
    }
}

// ---- LayerNorm over 1280, in-place fp32 + f16 copy ------------------------
__global__ __launch_bounds__(256) void k_ln(float* __restrict__ x,
                                            _Float16* __restrict__ xh,
                                            const float* __restrict__ gma,
                                            const float* __restrict__ bta)
{
    const int N = PLMD;
    __shared__ float rs[256], rs2[256];
    float* xr = x + (size_t)blockIdx.x * N;
    float s = 0.f, s2 = 0.f;
    for (int j = threadIdx.x; j < N; j += 256) { float v = xr[j]; s += v; s2 += v * v; }
    rs[threadIdx.x] = s; rs2[threadIdx.x] = s2;
    __syncthreads();
    for (int st = 128; st > 0; st >>= 1) {
        if (threadIdx.x < st) {
            rs[threadIdx.x]  += rs[threadIdx.x + st];
            rs2[threadIdx.x] += rs2[threadIdx.x + st];
        }
        __syncthreads();
    }
    float mu  = rs[0] * (1.0f / N);
    float var = rs2[0] * (1.0f / N) - mu * mu;
    float inv = rsqrtf(var + 1e-5f);
    for (int j = threadIdx.x; j < N; j += 256) {
        float v = (xr[j] - mu) * inv * gma[j] + bta[j];
        xr[j] = v;                                  // residual for final add
        xh[(size_t)blockIdx.x * N + j] = (_Float16)v;
    }
}

// ---------------------------------------------------------------------------
extern "C" void kernel_launch(void* const* d_in, const int* in_sizes, int n_in,
                              void* d_out, int out_size, void* d_ws, size_t ws_size,
                              hipStream_t stream)
{
    (void)in_sizes; (void)n_in; (void)out_size; (void)ws_size;
    const float* s_rep = (const float*)d_in[0];
    const float* plm   = (const float*)d_in[1];
    const float* Wq = (const float*)d_in[2];  const float* bq = (const float*)d_in[3];
    const float* Wk = (const float*)d_in[4];  const float* bk = (const float*)d_in[5];
    const float* Wv = (const float*)d_in[6];  const float* bv = (const float*)d_in[7];
    const float* Wo = (const float*)d_in[8];  const float* bo = (const float*)d_in[9];
    const float* lg = (const float*)d_in[10]; const float* lb = (const float*)d_in[11];
    const float* Wd = (const float*)d_in[12]; const float* bd = (const float*)d_in[13];
    const float* Wu = (const float*)d_in[14]; const float* bu = (const float*)d_in[15];

    // ---- workspace layout (manual, with reuse) ----
    char* ws  = (char*)d_ws;
    size_t off = 0;
    auto take = [&](size_t bytes) -> char* {
        char* p = ws + off; off += (bytes + 255) & ~(size_t)255; return p;
    };
    _Float16* whq  = (_Float16*)take((size_t)EMB * PLMD * 2);
    _Float16* whk  = (_Float16*)take((size_t)EMB * 128 * 2);
    _Float16* whv  = (_Float16*)take((size_t)EMB * 128 * 2);
    _Float16* who  = (_Float16*)take((size_t)PLMD * EMB * 2);
    _Float16* whd  = (_Float16*)take((size_t)FFD * PLMD * 2);
    _Float16* whu  = (_Float16*)take((size_t)PLMD * FFD * 2);
    _Float16* shh  = (_Float16*)take((size_t)ROWS * 128 * 2);
    _Float16* plmh = (_Float16*)take((size_t)ROWS * PLMD * 2);   // dead after Q gemm
    _Float16* qh   = (_Float16*)take((size_t)ROWS * EMB * 2);
    _Float16* khb  = (_Float16*)take((size_t)ROWS * EMB * 2);
    _Float16* vTb  = (_Float16*)take((size_t)ROWS * EMB * 2);
    char*     big  = take(3 * (size_t)ROWS * EMB * 4);           // qf/kf/vf f32 staging

    float* qf = (float*)big;
    float* kf = (float*)(big + (size_t)ROWS * EMB * 4);
    float* vf = (float*)(big + 2 * (size_t)ROWS * EMB * 4);
    // after k_rope, the staging region is recycled:
    float*    xbuf = (float*)big;                                       // 21.0 MB
    _Float16* xh   = (_Float16*)(big + (size_t)ROWS * PLMD * 4);        // 10.5 MB
    _Float16* hmid = (_Float16*)(big + (size_t)ROWS * PLMD * 4
                                     + (size_t)ROWS * PLMD * 2);        //  5.2 MB
    _Float16* valsh = plmh;   // reuse (plm f16 dead after the Q projection)

    float* outp    = (float*)d_out;
    float* attnout = outp + (size_t)ROWS * PLMD;   // tuple: (out, attention)

    auto cvt = [&](const float* s, _Float16* d, int n) {
        k_cvt<<<(n + 255) / 256, 256, 0, stream>>>(s, d, n);
    };
    // weight / activation conversions
    cvt(Wq, whq, EMB * PLMD);
    cvt(Wk, whk, EMB * 128);
    cvt(Wv, whv, EMB * 128);
    cvt(Wo, who, PLMD * EMB);
    cvt(Wd, whd, FFD * PLMD);
    cvt(Wu, whu, PLMD * FFD);
    cvt(plm, plmh, ROWS * PLMD);
    cvt(s_rep, shh, ROWS * 128);

    // QKV projections (fp32 staging for RoPE)
    k_gemm<<<dim3(EMB / 64, ROWS / 64), 256, 0, stream>>>(
        plmh, whq, bq, nullptr, qf, nullptr, ROWS, EMB, PLMD, 0);
    k_gemm<<<dim3(EMB / 64, ROWS / 64), 256, 0, stream>>>(
        shh, whk, bk, nullptr, kf, nullptr, ROWS, EMB, 128, 0);
    k_gemm<<<dim3(EMB / 64, ROWS / 64), 256, 0, stream>>>(
        shh, whv, bv, nullptr, vf, nullptr, ROWS, EMB, 128, 0);

    // RoPE + f16 relayout (q scaled by 1/8; v transposed)
    k_rope<<<(ROWS * EMB + 255) / 256, 256, 0, stream>>>(qf, kf, vf, qh, khb, vTb);

    // fused attention (writes attention fp32 to d_out, vals f16 to ws)
    k_attn<<<dim3(SLEN / 16, NHEADS, BATCH), 256, 0, stream>>>(
        qh, khb, vTb, attnout, valsh);

    // O-projection + residual(plm) -> xbuf
    k_gemm<<<dim3(PLMD / 64, ROWS / 64), 256, 0, stream>>>(
        valsh, who, bo, plm, xbuf, nullptr, ROWS, PLMD, EMB, 1);

    // LayerNorm (in-place fp32 residual + f16 copy for FFN)
    k_ln<<<ROWS, 256, 0, stream>>>(xbuf, xh, lg, lb);

    // FFN: GELU(x @ Wd^T + bd) -> hmid (f16)
    k_gemm<<<dim3(FFD / 64, ROWS / 64), 256, 0, stream>>>(
        xh, whd, bd, nullptr, nullptr, hmid, ROWS, FFD, PLMD, 2);

    // out = hmid @ Wu^T + bu + residual -> d_out
    k_gemm<<<dim3(PLMD / 64, ROWS / 64), 256, 0, stream>>>(
        hmid, whu, bu, xbuf, outp, nullptr, ROWS, PLMD, FFD, 1);
}
